// TMSSPDModel_60559038874133
// MI455X (gfx1250) — compile-verified, matrix operand order checked by reading, not run on hardware
//
#include <hip/hip_runtime.h>
#include <stdint.h>

typedef __attribute__((ext_vector_type(16))) __bf16        v16bf;
typedef __attribute__((ext_vector_type(8)))  float         v8f;
typedef __attribute__((ext_vector_type(8)))  unsigned int  v8u;
// TDM descriptor groups (6-arg builtin: v4u, v8i, v4i, v4i, v8i, i32)
typedef __attribute__((ext_vector_type(4)))  unsigned int  tdm_v4u;
typedef __attribute__((ext_vector_type(8)))  int           tdm_v8i;
typedef __attribute__((ext_vector_type(4)))  int           tdm_v4i;

#define B_SZ   2048
#define N_INST 8
#define N_FEAT 100
#define N_HID  40
#define NC     40
#define NM     41

// padded GEMM dims
#define KF 128   // pass-1 K (=F padded), 4 k-chunks of 32
#define NH 48    // pass-1 N (=H padded), 3 n-tiles of 16
#define KH 64    // pass-2 K (=H padded), 2 k-chunks
#define NF 112   // pass-2 N (=F padded), 7 n-tiles

#define FRAG_ELEMS 512   // 32 lanes * 16 bf16 per 32x16 B-fragment
#define NFRAG1 12        // 4 kc * 3 nt  -> 12 KB per (i,c)
#define NFRAG2 14        // 2 kc * 7 nt  -> 14 KB per (i,c)
#define STAGE_BYTES (NFRAG2 * FRAG_ELEMS * 2)   // 14336 B, max of the two passes

// CDNA5 16-bit A/B fragment element mapping (ISA 7.12.2):
// within a 32-K chunk, element kk lives in lane-half (kk>>3)&1,
// at packed index j = (kk&7) | ((kk&16)>>1).
__device__ __forceinline__ int frag_lane(int kk, int nn) { return nn + 16 * ((kk >> 3) & 1); }
__device__ __forceinline__ int frag_j(int kk)            { return (kk & 7) | ((kk & 16) >> 1); }

// ---------------------------------------------------------------------------
// TDM: 1-D contiguous DMA (Global -> LDS), nbytes multiple of 8.
// D# per ISA 8.3/8.4: count=1, type=2, data_size=8B, tensor_dim0=tile_dim0=n8,
// tensor_dim1=tile_dim1=1, no gather/iterate/pad, workgroup_mask=0.
// ---------------------------------------------------------------------------
__device__ __forceinline__ void tdm_load_1d(uint32_t lds_off, const void* gptr, uint32_t nbytes) {
    uint64_t ga = (uint64_t)(uintptr_t)gptr;
    uint32_t n8 = nbytes >> 3;
    tdm_v4u g0;
    g0.x = 1u;                                      // count=1 (valid), user mode
    g0.y = lds_off;                                 // lds_addr (bytes)
    g0.z = (uint32_t)ga;                            // global_addr[31:0]
    g0.w = (uint32_t)(ga >> 32) | (2u << 30);       // global_addr[56:32] | type=2
    tdm_v8i g1;
    g1[0] = (int)(3u << 16);                        // wg_mask=0 | data_size=3 (8B)
    g1[1] = (int)(n8 << 16);                        // bar_addr=0 | tensor_dim0[15:0]
    g1[2] = (int)((n8 >> 16) | (1u << 16));         // tensor_dim0[31:16] | tensor_dim1=1
    g1[3] = (int)(n8 << 16);                        // tensor_dim1[31:16]=0 | tile_dim0=n8
    g1[4] = 1;                                      // tile_dim1=1 | tile_dim2=0
    g1[5] = (int)n8;                                // tensor_dim0_stride[31:0]
    g1[6] = 0;                                      // stride0[47:32] | dim1_stride[15:0]
    g1[7] = 0;
    tdm_v4i z4 = {0, 0, 0, 0};
    tdm_v8i z8 = {0, 0, 0, 0, 0, 0, 0, 0};
    __builtin_amdgcn_tensor_load_to_lds(g0, g1, z4, z4, z8, 0);
}

__device__ __forceinline__ uint32_t lds_offset_of(const void* p) {
    // LDS aperture: flat addr[31:0] == byte offset within the workgroup's LDS.
    return (uint32_t)(uintptr_t)p;
}

// ---------------------------------------------------------------------------
// Pack kernel: W[i,c] = A[i,c] (FxM) @ Bp[i,c] (MxH); store as pre-swizzled
// bf16 WMMA B-fragments, both as W (K=F,N=H) and W^T (K=H,N=F).
// ---------------------------------------------------------------------------
__global__ __launch_bounds__(128)
void pack_weights(const float* __restrict__ A, const float* __restrict__ Bp,
                  __bf16* __restrict__ W1p, __bf16* __restrict__ Vp) {
    __shared__ __bf16 Ws[N_FEAT][N_HID];   // W[f][h]
    const int bc  = blockIdx.x;            // = i*NC + c
    const int tid = threadIdx.x;
    const float* Aic = A  + (size_t)bc * N_FEAT * NM;   // A[i,c,f,m]
    const float* Bic = Bp + (size_t)bc * NM * N_HID;    // Bp[i,c,m,h]

    for (int e = tid; e < N_FEAT * N_HID; e += 128) {
        int f = e / N_HID, h = e % N_HID;
        float acc = 0.f;
        for (int m = 0; m < NM; ++m)
            acc += Aic[f * NM + m] * Bic[m * N_HID + h];
        Ws[f][h] = (__bf16)acc;
    }
    __syncthreads();

    // pass-1 fragments: K = f (0..127), N = h (0..47)
    __bf16* w1 = W1p + (size_t)bc * NFRAG1 * FRAG_ELEMS;
    for (int e = tid; e < KF * NH; e += 128) {
        int k = e / NH, n = e % NH;
        int kc = k >> 5, kk = k & 31, nt = n >> 4, nn = n & 15;
        __bf16 v = (k < N_FEAT && n < N_HID) ? Ws[k][n] : (__bf16)0.0f;
        w1[(size_t)(kc * 3 + nt) * FRAG_ELEMS + frag_lane(kk, nn) * 16 + frag_j(kk)] = v;
    }
    // pass-2 fragments: K = h (0..63), N = f (0..111); V[h][f] = W[f][h]
    __bf16* vp = Vp + (size_t)bc * NFRAG2 * FRAG_ELEMS;
    for (int e = tid; e < KH * NF; e += 128) {
        int k = e / NF, n = e % NF;
        int kc = k >> 5, kk = k & 31, nt = n >> 4, nn = n & 15;
        __bf16 v = (k < N_HID && n < N_FEAT) ? Ws[n][k] : (__bf16)0.0f;
        vp[(size_t)(kc * 7 + nt) * FRAG_ELEMS + frag_lane(kk, nn) * 16 + frag_j(kk)] = v;
    }
}

// ---------------------------------------------------------------------------
// Fused main kernel: 4 wave32s per block, one 16-row b-tile each, same
// instance. Per component c, wave 0 DMAs the pre-swizzled fragment block
// into double-buffered LDS via the Tensor Data Mover; all 4 waves consume it
// (4x reduction of L2 fragment traffic vs per-wave loads).
//   h   = sum_c mask[b,c] * (x_tile @ W[i,c])       (K=128, 3 n-tiles)
//   out = relu( sum_c mask[b,c] * (h @ W[i,c]^T) )  (K=64,  7 n-tiles)
// ---------------------------------------------------------------------------
__global__ __launch_bounds__(128)
void tms_fused(const float* __restrict__ x, const float* __restrict__ mask,
               const __bf16* __restrict__ W1p, const __bf16* __restrict__ Vp,
               const float* __restrict__ bfin, float* __restrict__ out) {
    __shared__ __attribute__((aligned(32))) unsigned char stage[2][STAGE_BYTES]; // 28 KB
    __shared__ __bf16 xs[4][16][KF];   // 16 KB: per-wave x tile, bf16, padded K
    __shared__ float  hs[4][16][KH];   // 16 KB: per-wave h round-trip

    const int tid   = threadIdx.x;
    const int wave  = tid >> 5;
    const int l     = tid & 31;
    const int mrow  = l & 15;
    const int khalf = l >> 4;
    const int inst  = blockIdx.x >> 5;                 // 256 blocks: 32 groups x 8 inst
    const int bt    = (blockIdx.x & 31) * 4 + wave;    // b-tile of this wave

    const bool issuer = (wave == 0);                   // wave-uniform

    // zero hs (cols >= 48 stay zero as pass-2 K padding)
    for (int idx = l; idx < 16 * KH; idx += 32) ((float*)hs[wave])[idx] = 0.f;

    // load + convert this wave's x tile
    for (int idx = l; idx < 16 * KF; idx += 32) {
        int r = idx >> 7, col = idx & 127;
        float v = (col < N_FEAT)
                      ? x[((size_t)(bt * 16 + r) * N_INST + inst) * N_FEAT + col]
                      : 0.f;
        xs[wave][r][col] = (__bf16)v;
    }

    const __bf16* w1base = W1p + (size_t)(inst * NC) * NFRAG1 * FRAG_ELEMS;
    const __bf16* vpbase = Vp  + (size_t)(inst * NC) * NFRAG2 * FRAG_ELEMS;
    const uint32_t stage_off0 = lds_offset_of(&stage[0][0]);
    const uint32_t stage_off1 = lds_offset_of(&stage[1][0]);

    // prologue: stage pass-1 fragments for c=0
    if (issuer) tdm_load_1d(stage_off0, w1base, NFRAG1 * FRAG_ELEMS * 2);

    __syncthreads();   // xs visible (and orders hs zeroing)

    // build pass-1 A fragments (rows = b, K = f)
    v16bf a1[4];
#pragma unroll
    for (int kc = 0; kc < 4; ++kc)
#pragma unroll
        for (int j = 0; j < 16; ++j) {
            int K = kc * 32 + ((j < 8) ? j : j + 8) + 8 * khalf;
            a1[kc][j] = xs[wave][mrow][K];
        }

    v8f hacc[3] = {};
    const float* mk = mask + ((size_t)(bt * 16 + mrow) * N_INST + inst) * NC;

    for (int c = 0; c < NC; ++c) {
        if (issuer) {
            if (c + 1 < NC) {
                tdm_load_1d((c & 1) ? stage_off0 : stage_off1,
                            w1base + (size_t)(c + 1) * NFRAG1 * FRAG_ELEMS,
                            NFRAG1 * FRAG_ELEMS * 2);
                __builtin_amdgcn_s_wait_tensorcnt(1);   // oldest (buffer c) ready
            } else {
                __builtin_amdgcn_s_wait_tensorcnt(0);   // last buffer ready
            }
        }
        __syncthreads();                                // publish buffer
        const unsigned char* sb = stage[c & 1];
        unsigned int msel = (mk[c] != 0.f) ? ~0u : 0u;  // mask is binary 0/1
#pragma unroll
        for (int kc = 0; kc < 4; ++kc) {
            v16bf am = __builtin_bit_cast(v16bf, __builtin_bit_cast(v8u, a1[kc]) & msel);
#pragma unroll
            for (int nt = 0; nt < 3; ++nt) {
                v16bf bfrag = *(const v16bf*)(sb + (kc * 3 + nt) * (FRAG_ELEMS * 2) + l * 32);
                hacc[nt] = __builtin_amdgcn_wmma_f32_16x16x32_bf16(
                    false, am, false, bfrag, (short)0, hacc[nt], false, false);
            }
        }
        __syncthreads();   // all waves done with this buffer before it is reused
    }

    // spill h (C/D layout) to LDS, re-read in A layout
#pragma unroll
    for (int nt = 0; nt < 3; ++nt)
#pragma unroll
        for (int v = 0; v < 8; ++v)
            hs[wave][v + 8 * khalf][nt * 16 + (l & 15)] = hacc[nt][v];

    // prologue: stage pass-2 fragments for c=0 (both buffers idle here)
    if (issuer) tdm_load_1d(stage_off0, vpbase, NFRAG2 * FRAG_ELEMS * 2);

    __syncthreads();

    v16bf a2[2];
#pragma unroll
    for (int kc = 0; kc < 2; ++kc)
#pragma unroll
        for (int j = 0; j < 16; ++j) {
            int K = kc * 32 + ((j < 8) ? j : j + 8) + 8 * khalf;
            a2[kc][j] = (__bf16)hs[wave][mrow][K];
        }

    v8f oacc[7] = {};

    for (int c = 0; c < NC; ++c) {
        if (issuer) {
            if (c + 1 < NC) {
                tdm_load_1d((c & 1) ? stage_off0 : stage_off1,
                            vpbase + (size_t)(c + 1) * NFRAG2 * FRAG_ELEMS,
                            NFRAG2 * FRAG_ELEMS * 2);
                __builtin_amdgcn_s_wait_tensorcnt(1);
            } else {
                __builtin_amdgcn_s_wait_tensorcnt(0);
            }
        }
        __syncthreads();
        const unsigned char* sb = stage[c & 1];
        unsigned int msel = (mk[c] != 0.f) ? ~0u : 0u;
#pragma unroll
        for (int kc = 0; kc < 2; ++kc) {
            v16bf am = __builtin_bit_cast(v16bf, __builtin_bit_cast(v8u, a2[kc]) & msel);
#pragma unroll
            for (int nt = 0; nt < 7; ++nt) {
                v16bf bfrag = *(const v16bf*)(sb + (kc * 7 + nt) * (FRAG_ELEMS * 2) + l * 32);
                oacc[nt] = __builtin_amdgcn_wmma_f32_16x16x32_bf16(
                    false, am, false, bfrag, (short)0, oacc[nt], false, false);
            }
        }
        __syncthreads();
    }

    // epilogue: relu(out + b_final)
#pragma unroll
    for (int nt = 0; nt < 7; ++nt) {
        int f = nt * 16 + (l & 15);
        if (f < N_FEAT) {
            float bb = bfin[inst * N_FEAT + f];
#pragma unroll
            for (int v = 0; v < 8; ++v) {
                int b = bt * 16 + v + 8 * khalf;
                float val = oacc[nt][v] + bb;
                out[((size_t)b * N_INST + inst) * N_FEAT + f] = val > 0.f ? val : 0.f;
            }
        }
    }
}

extern "C" void kernel_launch(void* const* d_in, const int* in_sizes, int n_in,
                              void* d_out, int out_size, void* d_ws, size_t ws_size,
                              hipStream_t stream) {
    const float* x    = (const float*)d_in[0];   // (B, I, F)
    const float* mask = (const float*)d_in[1];   // (B, I, C)
    const float* A    = (const float*)d_in[2];   // (I, C, F, M)
    const float* Bp   = (const float*)d_in[3];   // (I, C, M, H)
    const float* bfin = (const float*)d_in[4];   // (I, F)
    float* out = (float*)d_out;

    // workspace: packed bf16 fragments
    __bf16* W1p = (__bf16*)d_ws;
    __bf16* Vp  = (__bf16*)((char*)d_ws +
                            (size_t)N_INST * NC * NFRAG1 * FRAG_ELEMS * sizeof(__bf16));

    pack_weights<<<N_INST * NC, 128, 0, stream>>>(A, Bp, W1p, Vp);
    tms_fused<<<(B_SZ / 16) * N_INST, 128, 0, stream>>>(x, mask, W1p, Vp, bfin, out);
}
